// VisionTransformer_9869834847061
// MI455X (gfx1250) — matrix operand order repaired and checked by reading, not compiled, not run on hardware
//
#include <hip/hip_runtime.h>
#include <hip/hip_bf16.h>

// ---------------------------------------------------------------------------
// ViT forward (DEPTH=4, DIM=512, HEADS=8, hd=64, HID=1024, B=64, NTOK=197)
// bf16 WMMA GEMMs (v_wmma_f32_16x16x32_bf16). All B operands pre-transposed
// to [N][K] so staging is pure b128 loads/stores. 128x64 block tile, 32x32
// per wave (4 WMMAs/step), double-buffered LDS, fp32 epilogues.
// ---------------------------------------------------------------------------

#define BATCH   64
#define NTOK    197
#define DIMX    512
#define HEADSX  8
#define HDX     64
#define HIDX    1024
#define NPATCH  196
#define PATCHK  768                   // 3*16*16
#define MTOK    (BATCH * NTOK)        // 12608
#define MPATCH  (BATCH * NPATCH)      // 12544 (multiple of 128)
#define SLD     224                   // padded score/prob row stride
#define VROWS   224                   // padded V K-extent: 7*32

typedef __attribute__((ext_vector_type(16))) __bf16 v16bf;
typedef __attribute__((ext_vector_type(8)))  float  v8f;
typedef __attribute__((ext_vector_type(4)))  unsigned int u32x4;

struct Pair16 { u32x4 lo, hi; };

__device__ __forceinline__ v16bf make_frag(u32x4 lo, u32x4 hi) {
  Pair16 p{lo, hi};
  return __builtin_bit_cast(v16bf, p);
}

__device__ __forceinline__ unsigned short f2bf(float f) {
  unsigned int u = __builtin_bit_cast(unsigned int, f);
  u += 0x7FFFu + ((u >> 16) & 1u);          // round-to-nearest-even
  return (unsigned short)(u >> 16);
}

#define WMMA_BF16(A_, B_, C_) \
  __builtin_amdgcn_wmma_f32_16x16x32_bf16(false, (A_), false, (B_), (short)0, (C_), false, false)

// ---------------------------------------------------------------------------
// Batched bf16 GEMM: C[z](MxN) = A[z](M x K, row-major, lda) *
//                                Bt[z](N x K, row-major, ldb)^T  (+bias)(+epi)
//   EPI: 0 f32 | 1 f32 + residual | 2 bf16 | 3 exact-GELU -> bf16
//   Batch C offset = (z/Hdiv)*sC1 + (z%Hdiv)*sC2   (head-merge epilogue)
// No load-side guards: caller guarantees over-reads (< 1 tile) stay inside
// the workspace and K-padding regions multiply by zero.
// ---------------------------------------------------------------------------
template<int EPI>
__global__ __launch_bounds__(256, 2)
void gemm_bf16_wmma(const unsigned short* __restrict__ A,
                    const unsigned short* __restrict__ Bt,
                    const float* __restrict__ bias,
                    const float* resid,
                    void* Cout,
                    int M, int N, int K,
                    int lda, int ldb, int ldc,
                    long long sA, long long sB,
                    long long sC1, long long sC2, int Hdiv)
{
  __shared__ __align__(16) unsigned short As[2][128 * 32];  // [m][k]
  __shared__ __align__(16) unsigned short Bs[2][64 * 32];   // [n][k]

  const int z = blockIdx.z;
  A  += (long long)z * sA;
  Bt += (long long)z * sB;
  const long long coff = (long long)(z / Hdiv) * sC1 + (long long)(z % Hdiv) * sC2;

  const int row0 = blockIdx.y * 128;
  const int col0 = blockIdx.x * 64;
  const int tid  = threadIdx.x;
  const int lane = tid & 31;
  const int wave = tid >> 5;      // 8 waves
  const int wm   = wave >> 1;     // 0..3 : 32-row strip
  const int wn   = wave & 1;      // 0..1 : 32-col strip
  const int lr   = lane & 15;
  const int lh   = lane >> 4;

  // staging maps: A = 32B/thread (2 x b128), B = 16B/thread (1 x b128)
  const int ar  = tid >> 1;        // 0..127
  const int akb = (tid & 1) * 16;  // 0 / 16
  const int br  = tid >> 2;        // 0..63
  const int bkb = (tid & 3) * 8;

  v8f c00 = {}, c01 = {}, c10 = {}, c11 = {};

  const int nSteps = (K + 31) >> 5;
  const unsigned short* Arow = A  + (long long)(row0 + ar) * lda + akb;
  const unsigned short* Brow = Bt + (long long)(col0 + br) * ldb + bkb;

  u32x4 ra0 = *(const u32x4*)(Arow);
  u32x4 ra1 = *(const u32x4*)(Arow + 8);
  u32x4 rb  = *(const u32x4*)(Brow);
  *(u32x4*)&As[0][ar * 32 + akb]     = ra0;
  *(u32x4*)&As[0][ar * 32 + akb + 8] = ra1;
  *(u32x4*)&Bs[0][br * 32 + bkb]     = rb;
  __syncthreads();

  for (int s = 0; s < nSteps; ++s) {
    const int cur = s & 1;
    if (s + 1 < nSteps) {                       // uniform prefetch branch
      const int k1 = (s + 1) << 5;
      ra0 = *(const u32x4*)(Arow + k1);
      ra1 = *(const u32x4*)(Arow + k1 + 8);
      rb  = *(const u32x4*)(Brow + k1);
    }
    {
      const int a0b = (wm * 32 + lr) * 32 + lh * 8;
      const int a1b = a0b + 16 * 32;
      const v16bf a0 = make_frag(*(const u32x4*)&As[cur][a0b],
                                 *(const u32x4*)&As[cur][a0b + 16]);
      const v16bf a1 = make_frag(*(const u32x4*)&As[cur][a1b],
                                 *(const u32x4*)&As[cur][a1b + 16]);
      const int b0b = (wn * 32 + lr) * 32 + lh * 8;
      const int b1b = b0b + 16 * 32;
      const v16bf b0 = make_frag(*(const u32x4*)&Bs[cur][b0b],
                                 *(const u32x4*)&Bs[cur][b0b + 16]);
      const v16bf b1 = make_frag(*(const u32x4*)&Bs[cur][b1b],
                                 *(const u32x4*)&Bs[cur][b1b + 16]);
      c00 = WMMA_BF16(a0, b0, c00);
      c01 = WMMA_BF16(a0, b1, c01);
      c10 = WMMA_BF16(a1, b0, c10);
      c11 = WMMA_BF16(a1, b1, c11);
    }
    if (s + 1 < nSteps) {
      const int nxt = cur ^ 1;
      *(u32x4*)&As[nxt][ar * 32 + akb]     = ra0;
      *(u32x4*)&As[nxt][ar * 32 + akb + 8] = ra1;
      *(u32x4*)&Bs[nxt][br * 32 + bkb]     = rb;
    }
    __syncthreads();
  }

  // epilogue: C/D layout — lane half selects M 0-7 / 8-15, N = lane&15
#pragma unroll
  for (int sub = 0; sub < 2; ++sub) {
#pragma unroll
    for (int rr = 0; rr < 8; ++rr) {
      const int m = row0 + wm * 32 + sub * 16 + lh * 8 + rr;
      if (m >= M) continue;
#pragma unroll
      for (int half = 0; half < 2; ++half) {
        const int n = col0 + wn * 32 + half * 16 + lr;
        if (n >= N) continue;
        float val = sub ? (half ? c11[rr] : c10[rr])
                        : (half ? c01[rr] : c00[rr]);
        if (bias) val += bias[n];
        const long long off = coff + (long long)m * ldc + n;
        if (EPI == 0) {
          ((float*)Cout)[off] = val;
        } else if (EPI == 1) {
          ((float*)Cout)[off] = val + resid[off];
        } else if (EPI == 2) {
          ((unsigned short*)Cout)[off] = f2bf(val);
        } else {
          const float g = 0.5f * val * (1.0f + erff(val * 0.70710678118654752f));
          ((unsigned short*)Cout)[off] = f2bf(g);
        }
      }
    }
  }
}

// --------------------------- LayerNorm (row = 512) -------------------------
template<bool OUTF32>
__global__ __launch_bounds__(256)
void layernorm_k(const float* __restrict__ x, const float* __restrict__ w,
                 const float* __restrict__ b, void* out)
{
  __shared__ float s1[256], s2[256];
  const long long row = blockIdx.x;
  const float* xr = x + row * DIMX;
  const int t = threadIdx.x;
  const float v0 = xr[t], v1 = xr[t + 256];
  s1[t] = v0 + v1;
  s2[t] = v0 * v0 + v1 * v1;
  __syncthreads();
  for (int s = 128; s > 0; s >>= 1) {
    if (t < s) { s1[t] += s1[t + s]; s2[t] += s2[t + s]; }
    __syncthreads();
  }
  const float mean = s1[0] * (1.0f / DIMX);
  const float var  = s2[0] * (1.0f / DIMX) - mean * mean;
  const float rstd = rsqrtf(var + 1e-5f);
  const float y0 = (v0 - mean) * rstd * w[t] + b[t];
  const float y1 = (v1 - mean) * rstd * w[t + 256] + b[t + 256];
  if (OUTF32) {
    float* o = (float*)out + row * DIMX;
    o[t] = y0; o[t + 256] = y1;
  } else {
    unsigned short* o = (unsigned short*)out + row * DIMX;
    o[t] = f2bf(y0); o[t + 256] = f2bf(y1);
  }
}

// ------------------------- softmax over one score row ----------------------
__global__ __launch_bounds__(256)
void softmax_k(const float* __restrict__ S, unsigned short* __restrict__ P)
{
  __shared__ float red[256];
  const long long row = blockIdx.x;                   // z*197 + m
  const float* sr = S + row * SLD;
  unsigned short* pr = P + row * SLD;
  const int t = threadIdx.x;
  const float v = (t < NTOK) ? sr[t] : -3.0e38f;
  red[t] = v; __syncthreads();
  for (int s = 128; s > 0; s >>= 1) {
    if (t < s) red[t] = fmaxf(red[t], red[t + s]);
    __syncthreads();
  }
  const float mx = red[0];
  __syncthreads();
  const float e = (t < NTOK) ? expf(v - mx) : 0.0f;
  red[t] = e; __syncthreads();
  for (int s = 128; s > 0; s >>= 1) {
    if (t < s) red[t] += red[t + s];
    __syncthreads();
  }
  const float inv = 1.0f / red[0];
  if (t < NTOK)     pr[t] = f2bf(e * inv);
  else if (t < SLD) pr[t] = 0;          // zero K-padding for P@V GEMM
}

// --------------------------- small helper kernels --------------------------
// in: [D][K][N] f32 row-major  ->  out: [D][N][K] bf16  (weight transpose)
__global__ void cvtT_bf16_k(const float* __restrict__ in,
                            unsigned short* __restrict__ out,
                            int K, int N, int total)
{
  const int idx = blockIdx.x * 256 + threadIdx.x;
  if (idx >= total) return;
  const int n = idx % N;
  const int k = (idx / N) % K;
  const int d = idx / (N * K);
  out[((long long)d * N + n) * K + k] = f2bf(in[idx]);
}

__global__ void patchify_k(const float* __restrict__ x,
                           unsigned short* __restrict__ p, int total)
{
  const int idx = blockIdx.x * 256 + threadIdx.x;
  if (idx >= total) return;
  const int f = idx % PATCHK;
  const int t = (idx / PATCHK) % NPATCH;
  const int b = idx / (PATCHK * NPATCH);
  const int c = f >> 8, ph = (f >> 4) & 15, pw = f & 15;
  const int gy = t / 14, gx = t % 14;
  const float v = x[(((long long)b * 3 + c) * 224 + gy * 16 + ph) * 224
                    + gx * 16 + pw];
  p[idx] = f2bf(v);
}

__global__ void assemble_k(const float* __restrict__ tok,
                           const float* __restrict__ cls,
                           const float* __restrict__ pos,
                           float* __restrict__ h, int total)
{
  const int idx = blockIdx.x * 256 + threadIdx.x;
  if (idx >= total) return;
  const int d = idx & (DIMX - 1);
  const int n = (idx >> 9) % NTOK;
  const int b = idx / (NTOK * DIMX);
  const float base = (n == 0) ? cls[d]
                              : tok[((long long)b * NPATCH + (n - 1)) * DIMX + d];
  h[idx] = base + pos[n * DIMX + d];
}

// qkv f32 [B*N][3*DIM] -> q,k bf16 [B*H][197][64] ; V^T bf16 [B*H][64][224]
__global__ void qkv_split_k(const float* __restrict__ qkv,
                            unsigned short* __restrict__ q,
                            unsigned short* __restrict__ k,
                            unsigned short* __restrict__ vt, int total)
{
  const int idx = blockIdx.x * 256 + threadIdx.x;
  if (idx >= total) return;
  const int col = idx % (3 * DIMX);
  const int row = idx / (3 * DIMX);
  const int b = row / NTOK, n = row % NTOK;
  const int s = col / DIMX, rem = col % DIMX;
  const int hh = rem >> 6, d = rem & 63;
  const float val = qkv[idx];
  const long long zq = (long long)b * HEADSX + hh;
  if (s == 0)      q[(zq * NTOK + n) * HDX + d] = f2bf(val * 0.125f);
  else if (s == 1) k[(zq * NTOK + n) * HDX + d] = f2bf(val);
  else             vt[(zq * HDX + d) * VROWS + n] = f2bf(val); // transposed
}

__global__ void zero_vpad_k(unsigned short* __restrict__ vt, int total)
{
  const int idx = blockIdx.x * 256 + threadIdx.x;   // (B*H*64) * 27
  if (idx >= total) return;
  const int r   = idx % (VROWS - NTOK);
  const int drow = idx / (VROWS - NTOK);            // 0 .. B*H*64-1
  vt[(long long)drow * VROWS + NTOK + r] = 0;
}

// ---------------------------------------------------------------------------
extern "C" void kernel_launch(void* const* d_in, const int* in_sizes, int n_in,
                              void* d_out, int out_size, void* d_ws, size_t ws_size,
                              hipStream_t stream)
{
  (void)in_sizes; (void)n_in; (void)out_size; (void)ws_size;
  const float* x      = (const float*)d_in[0];
  const float* conv_w = (const float*)d_in[1];
  const float* conv_b = (const float*)d_in[2];
  const float* clstk  = (const float*)d_in[3];
  const float* pos    = (const float*)d_in[4];
  const float* ln1_w  = (const float*)d_in[5];
  const float* ln1_b  = (const float*)d_in[6];
  const float* qkv_w  = (const float*)d_in[7];
  const float* qkv_b  = (const float*)d_in[8];
  const float* proj_w = (const float*)d_in[9];
  const float* proj_b = (const float*)d_in[10];
  const float* ln2_w  = (const float*)d_in[11];
  const float* ln2_b  = (const float*)d_in[12];
  const float* fc1_w  = (const float*)d_in[13];
  const float* fc1_b  = (const float*)d_in[14];
  const float* fc2_w  = (const float*)d_in[15];
  const float* fc2_b  = (const float*)d_in[16];
  const float* norm_w = (const float*)d_in[17];
  const float* norm_b = (const float*)d_in[18];

  // ---- workspace carve-up (256B aligned) ----
  char* base = (char*)d_ws;
  size_t off = 0;
  auto alloc = [&](size_t bytes) -> void* {
    void* p = base + off;
    off = (off + bytes + 255) & ~(size_t)255;
    return p;
  };
  const int ZBH = BATCH * HEADSX;               // 512
  unsigned short* wcT   = (unsigned short*)alloc((size_t)DIMX * PATCHK * 2);
  unsigned short* wqkvT = (unsigned short*)alloc((size_t)4 * 3 * DIMX * DIMX * 2);
  unsigned short* wprojT= (unsigned short*)alloc((size_t)4 * DIMX * DIMX * 2);
  unsigned short* wfc1T = (unsigned short*)alloc((size_t)4 * HIDX * DIMX * 2);
  unsigned short* wfc2T = (unsigned short*)alloc((size_t)4 * DIMX * HIDX * 2);
  unsigned short* patches = (unsigned short*)alloc((size_t)MPATCH * PATCHK * 2);
  float*          tok   = (float*)alloc((size_t)MPATCH * DIMX * 4);
  float*          h     = (float*)alloc((size_t)MTOK * DIMX * 4);
  unsigned short* y     = (unsigned short*)alloc((size_t)MTOK * DIMX * 2);
  float*          qkvf  = (float*)alloc((size_t)MTOK * 3 * DIMX * 4);
  unsigned short* qb    = (unsigned short*)alloc((size_t)ZBH * NTOK * HDX * 2);
  unsigned short* kb    = (unsigned short*)alloc((size_t)ZBH * NTOK * HDX * 2);
  unsigned short* vt    = (unsigned short*)alloc((size_t)ZBH * HDX * VROWS * 2);
  float*          scores= (float*)alloc((size_t)ZBH * NTOK * SLD * 4);
  unsigned short* probs = (unsigned short*)alloc((size_t)ZBH * NTOK * SLD * 2);
  unsigned short* ob    = (unsigned short*)alloc((size_t)MTOK * DIMX * 2);
  unsigned short* g     = (unsigned short*)alloc((size_t)MTOK * HIDX * 2);
  (void)alloc((size_t)512 * 1024);              // tail slack for over-reads

  const dim3 blk(256);
  auto nblk = [](int n) { return dim3((unsigned)((n + 255) / 256)); };
  const int MY = (MTOK + 127) / 128;            // 99 row-tiles for token GEMMs

  // ---- weights -> bf16 transposed [N][K]; zero V K-padding ----
  cvtT_bf16_k<<<nblk(PATCHK * DIMX), blk, 0, stream>>>(conv_w, wcT, PATCHK, DIMX, PATCHK * DIMX);
  cvtT_bf16_k<<<nblk(4 * DIMX * 3 * DIMX), blk, 0, stream>>>(qkv_w, wqkvT, DIMX, 3 * DIMX, 4 * DIMX * 3 * DIMX);
  cvtT_bf16_k<<<nblk(4 * DIMX * DIMX), blk, 0, stream>>>(proj_w, wprojT, DIMX, DIMX, 4 * DIMX * DIMX);
  cvtT_bf16_k<<<nblk(4 * DIMX * HIDX), blk, 0, stream>>>(fc1_w, wfc1T, DIMX, HIDX, 4 * DIMX * HIDX);
  cvtT_bf16_k<<<nblk(4 * HIDX * DIMX), blk, 0, stream>>>(fc2_w, wfc2T, HIDX, DIMX, 4 * HIDX * DIMX);
  zero_vpad_k<<<nblk(ZBH * HDX * (VROWS - NTOK)), blk, 0, stream>>>(
      vt, ZBH * HDX * (VROWS - NTOK));

  // ---- patch embed ----
  patchify_k<<<nblk(MPATCH * PATCHK), blk, 0, stream>>>(x, patches, MPATCH * PATCHK);
  gemm_bf16_wmma<0><<<dim3(DIMX / 64, MPATCH / 128, 1), blk, 0, stream>>>(
      patches, wcT, conv_b, nullptr, tok,
      MPATCH, DIMX, PATCHK, PATCHK, PATCHK, DIMX, 0, 0, 0, 0, 1);
  assemble_k<<<nblk(MTOK * DIMX), blk, 0, stream>>>(tok, clstk, pos, h, MTOK * DIMX);

  // ---- transformer blocks ----
  for (int i = 0; i < 4; ++i) {
    layernorm_k<false><<<dim3(MTOK), blk, 0, stream>>>(
        h, ln1_w + i * DIMX, ln1_b + i * DIMX, y);

    gemm_bf16_wmma<0><<<dim3(3 * DIMX / 64, MY, 1), blk, 0, stream>>>(
        y, wqkvT + (size_t)i * 3 * DIMX * DIMX, qkv_b + i * 3 * DIMX, nullptr, qkvf,
        MTOK, 3 * DIMX, DIMX, DIMX, DIMX, 3 * DIMX, 0, 0, 0, 0, 1);

    qkv_split_k<<<nblk(MTOK * 3 * DIMX), blk, 0, stream>>>(
        qkvf, qb, kb, vt, MTOK * 3 * DIMX);

    // S = (Q/8) * K^T  (K already [n][k] naturally) : batched over B*H
    gemm_bf16_wmma<0><<<dim3(4, 2, ZBH), blk, 0, stream>>>(
        qb, kb, nullptr, nullptr, scores,
        NTOK, NTOK, HDX, HDX, HDX, SLD,
        (long long)NTOK * HDX, (long long)NTOK * HDX,
        (long long)NTOK * SLD, 0, 1);

    softmax_k<<<dim3(ZBH * NTOK), blk, 0, stream>>>(scores, probs);

    // O = P * V  (V stored transposed [64][224], zero K-padding both sides)
    gemm_bf16_wmma<2><<<dim3(1, 2, ZBH), blk, 0, stream>>>(
        probs, vt, nullptr, nullptr, ob,
        NTOK, HDX, NTOK, SLD, VROWS, DIMX,
        (long long)NTOK * SLD, (long long)HDX * VROWS,
        (long long)NTOK * DIMX, (long long)HDX, HEADSX);

    // h = h + O*Wproj + b
    gemm_bf16_wmma<1><<<dim3(DIMX / 64, MY, 1), blk, 0, stream>>>(
        ob, wprojT + (size_t)i * DIMX * DIMX, proj_b + i * DIMX, h, h,
        MTOK, DIMX, DIMX, DIMX, DIMX, DIMX, 0, 0, 0, 0, 1);

    layernorm_k<false><<<dim3(MTOK), blk, 0, stream>>>(
        h, ln2_w + i * DIMX, ln2_b + i * DIMX, y);

    // g = gelu(y*W1 + b1)
    gemm_bf16_wmma<3><<<dim3(HIDX / 64, MY, 1), blk, 0, stream>>>(
        y, wfc1T + (size_t)i * HIDX * DIMX, fc1_b + i * HIDX, nullptr, g,
        MTOK, HIDX, DIMX, DIMX, DIMX, HIDX, 0, 0, 0, 0, 1);

    // h = h + g*W2 + b2
    gemm_bf16_wmma<1><<<dim3(DIMX / 64, MY, 1), blk, 0, stream>>>(
        g, wfc2T + (size_t)i * DIMX * HIDX, fc2_b + i * DIMX, h, h,
        MTOK, DIMX, HIDX, HIDX, HIDX, DIMX, 0, 0, 0, 0, 1);
  }

  // ---- final LN -> d_out (f32) ----
  layernorm_k<true><<<dim3(MTOK), blk, 0, stream>>>(h, norm_w, norm_b, d_out);
}
